// RLALinear_34789235098055
// MI455X (gfx1250) — matrix operand order repaired and checked by reading, not compiled, not run on hardware
//
#include <hip/hip_runtime.h>

// RLALinear (randomized sketch linear) for MI455X / gfx1250, wave32.
//   x[8,4096,1024] f32, W[1024,1024] f32, bias[1024] f32 -> out[8,4096,1024] f32
// Pipeline (all matmuls via v_wmma_f32_16x16x32_bf16, f32 accumulate):
//   k1: Sk  [B,din,k]  bf16 +-1   (threefry2x32-20, key 42)      -> ws
//   k2: SB  [B,k,dout] bf16 = Sk^T @ W^T                         -> ws
//   k3: AS  [B,S,k]    bf16 = (x @ Sk) / k                       -> ws
//   k4: out [B,S,dout] f32  = AS @ SB + bias
// HBM-bound (~268MB compulsory traffic @ 23.3TB/s ~= 11.5us floor); the 12MB of
// bf16 intermediates stay L2-resident (192MB L2). GEMM kernels use double-
// buffered LDS tiles (1 barrier/K-step, global loads overlapped with WMMA) and
// hoist all 8 B fragments into VGPRs so the 8 WMMAs issue back-to-back.

typedef __bf16 bf16_t;
typedef __bf16 v4bf  __attribute__((ext_vector_type(4)));
typedef __bf16 v8bf  __attribute__((ext_vector_type(8)));
typedef __bf16 v16bf __attribute__((ext_vector_type(16)));
typedef float  v8f   __attribute__((ext_vector_type(8)));

static constexpr int NB   = 8;     // batches
static constexpr int S    = 4096;  // sequence
static constexpr int DIN  = 1024;
static constexpr int DOUT = 1024;
static constexpr int KS   = 128;   // sketch size
static constexpr int KC   = 32;    // K-chunk per WMMA
static constexpr int LDA  = 40;    // A tile LDS row stride (halves): 32 + 8 pad -> conflict-free b128
static constexpr int LDB  = 136;   // B tile LDS row stride (halves): 128 + 8 pad

// ---------------- WMMA fragment loads (layouts per ISA 7.12.2) ----------------

// A 16x32 bf16, row-major LDS tile with stride LDA halves.
// element e: K = (e/8)*16 + (lane/16)*8 + (e%8); row = lane%16.
static __device__ __forceinline__ v16bf frag_a(const bf16_t* a_base, int lane) {
  const bf16_t* p = a_base + (lane & 15) * LDA + ((lane >> 4) << 3);
  v8bf lo = *(const v8bf*)(p);
  v8bf hi = *(const v8bf*)(p + 16);
  return __builtin_shufflevector(lo, hi, 0,1,2,3,4,5,6,7,8,9,10,11,12,13,14,15);
}

// B 32x16 bf16: N column = lane%16; element e: K = (e/8)*16 + (lane/16)*8 + (e%8); row-major LDS tile stride LDB.
static __device__ __forceinline__ v16bf frag_b(const bf16_t* b_tile, int lane, int n0) {
  const bf16_t* p = b_tile + ((lane >> 4) << 3) * LDB + n0 + (lane & 15);
  v16bf r;
#pragma unroll
  for (int e = 0; e < 8; ++e) {
    r[e]     = p[e * LDB];
    r[8 + e] = p[(16 + e) * LDB];
  }
  return r;
}

static __device__ __forceinline__ v8f wmma_bf16(v16bf a, v16bf b, v8f c) {
  return __builtin_amdgcn_wmma_f32_16x16x32_bf16(false, a, false, b, (short)0, c, false, false);
}

// ---------------- kernel 1: Rademacher sketch via threefry2x32-20 ----------------

static __device__ __forceinline__ unsigned rotl32(unsigned v, unsigned n) {
  return (v << n) | (v >> (32u - n));
}

static __device__ __forceinline__ void threefry2x32(unsigned c0, unsigned c1,
                                                    unsigned& o0, unsigned& o1) {
  const unsigned k0 = 0u, k1 = 42u;                 // jax.random.key(42)
  unsigned ks[3] = {k0, k1, 0x1BD11BDAu ^ k0 ^ k1};
  const unsigned R[8] = {13u,15u,26u,6u,17u,29u,16u,24u};
  unsigned x0 = c0 + ks[0], x1 = c1 + ks[1];
#pragma unroll
  for (int i = 0; i < 20; ++i) {
    x0 += x1; x1 = rotl32(x1, R[i & 7]); x1 ^= x0;
    if ((i & 3) == 3) {
      unsigned s = (unsigned)(i / 4) + 1u;
      x0 += ks[s % 3]; x1 += ks[(s + 1u) % 3] + s;
    }
  }
  o0 = x0; o1 = x1;
}

__global__ __launch_bounds__(256) void rla_gen_sketch(bf16_t* __restrict__ sk) {
  // one thread -> 64 consecutive +-1 values (one threefry call = 64 bits)
  unsigned chunk = blockIdx.x * 256u + threadIdx.x;   // NB*DIN*KS/64 = 16384 chunks
  unsigned r0, r1;
  threefry2x32(chunk, 0u, r0, r1);
  bf16_t* dst = sk + (size_t)chunk * 64;
#pragma unroll
  for (int j = 0; j < 32; ++j) {
    dst[j]      = ((r0 >> j) & 1u) ? (bf16_t)1.0f : (bf16_t)-1.0f;
    dst[32 + j] = ((r1 >> j) & 1u) ? (bf16_t)1.0f : (bf16_t)-1.0f;
  }
}

// ---------------- kernel 2: SB[b] = Sk[b]^T @ W^T  (128 x 1024, K=1024) ----------------
// A[m][ki] = Sk[b][ki][m] (transposed stage), B[ki][n] = W[n][ki] (transposed + cvt)

__global__ __launch_bounds__(256) void rla_sb(const bf16_t* __restrict__ sk,
                                              const float* __restrict__ w,
                                              bf16_t* __restrict__ sb) {
  __shared__ __align__(16) bf16_t As[2][128 * LDA];
  __shared__ __align__(16) bf16_t Bs[2][KC * LDB];
  const int tid = threadIdx.x, wave = tid >> 5, lane = tid & 31;
  const int n0blk = blockIdx.x * 128;               // 8 N-blocks
  const int b     = blockIdx.y;                     // 8 batches

  v8bf   areg[2];
  float4 breg[4];
  auto load_tile = [&](int k0) {
#pragma unroll
    for (int i = 0; i < 2; ++i) {
      int idx = tid + i * 256;                      // 512 8-half chunks of Sk
      int ir = idx >> 4, mc = (idx & 15) << 3;
      areg[i] = *(const v8bf*)(sk + ((size_t)b * DIN + k0 + ir) * KS + mc);
    }
#pragma unroll
    for (int i = 0; i < 4; ++i) {
      int idx = tid + i * 256;                      // 1024 float4 chunks of W
      int nr = idx >> 3, c4 = (idx & 7) << 2;
      breg[i] = *(const float4*)(w + ((size_t)(n0blk + nr)) * DIN + k0 + c4);
    }
  };
  auto store_tile = [&](int buf) {
#pragma unroll
    for (int i = 0; i < 2; ++i) {                   // transpose Sk chunk -> As
      int idx = tid + i * 256;
      int ir = idx >> 4, mc = (idx & 15) << 3;
#pragma unroll
      for (int j = 0; j < 8; ++j) As[buf][(mc + j) * LDA + ir] = areg[i][j];
    }
#pragma unroll
    for (int i = 0; i < 4; ++i) {                   // transpose + cvt W chunk -> Bs
      int idx = tid + i * 256;
      int nr = idx >> 3, c4 = (idx & 7) << 2;
      Bs[buf][(c4 + 0) * LDB + nr] = (bf16_t)breg[i].x;
      Bs[buf][(c4 + 1) * LDB + nr] = (bf16_t)breg[i].y;
      Bs[buf][(c4 + 2) * LDB + nr] = (bf16_t)breg[i].z;
      Bs[buf][(c4 + 3) * LDB + nr] = (bf16_t)breg[i].w;
    }
  };

  v8f acc[8];
#pragma unroll
  for (int t = 0; t < 8; ++t) acc[t] = (v8f)0.0f;

  load_tile(0);
  store_tile(0);
  __syncthreads();
  int cur = 0;
  for (int k0 = 0; k0 < DIN; k0 += KC) {
    const bool more = (k0 + KC) < DIN;
    if (more) load_tile(k0 + KC);                   // global loads overlap WMMA
    v16bf a = frag_a(&As[cur][0] + wave * 16 * LDA, lane);
    v16bf bfr[8];
#pragma unroll
    for (int nt = 0; nt < 8; ++nt) bfr[nt] = frag_b(&Bs[cur][0], lane, nt * 16);
#pragma unroll
    for (int nt = 0; nt < 8; ++nt) acc[nt] = wmma_bf16(a, bfr[nt], acc[nt]);
    if (more) {
      store_tile(cur ^ 1);
      __syncthreads();
      cur ^= 1;
    }
  }
  // epilogue: C layout (M = r + 8*(lane/16), N = lane%16)
#pragma unroll
  for (int nt = 0; nt < 8; ++nt) {
    int col = n0blk + nt * 16 + (lane & 15);
#pragma unroll
    for (int r = 0; r < 8; ++r) {
      int row = wave * 16 + r + ((lane >> 4) << 3);
      sb[((size_t)b * KS + row) * DOUT + col] = (bf16_t)acc[nt][r];
    }
  }
}

// ---------------- kernel 3: AS[b] = x[b] @ Sk[b] / 128  (4096 x 128, K=1024) ----------------

__global__ __launch_bounds__(256) void rla_as(const float* __restrict__ x,
                                              const bf16_t* __restrict__ sk,
                                              bf16_t* __restrict__ as_out) {
  __shared__ __align__(16) bf16_t As[2][128 * LDA];
  __shared__ __align__(16) bf16_t Bs[2][KC * LDB];
  const int tid = threadIdx.x, wave = tid >> 5, lane = tid & 31;
  const int b  = blockIdx.x >> 5;                   // 256 blocks = 8 * 32
  const int m0 = (blockIdx.x & 31) * 128;

  float4 areg[4];
  v8bf   breg[2];
  auto load_tile = [&](int k0) {
#pragma unroll
    for (int i = 0; i < 4; ++i) {
      int idx = tid + i * 256;                      // 1024 float4 chunks of x
      int row = idx >> 3, c4 = (idx & 7) << 2;
      areg[i] = *(const float4*)(x + ((size_t)b * S + m0 + row) * DIN + k0 + c4);
    }
#pragma unroll
    for (int i = 0; i < 2; ++i) {
      int idx = tid + i * 256;                      // 512 8-half chunks of Sk
      int row = idx >> 4, c8 = (idx & 15) << 3;
      breg[i] = *(const v8bf*)(sk + ((size_t)b * DIN + k0 + row) * KS + c8);
    }
  };
  auto store_tile = [&](int buf) {
#pragma unroll
    for (int i = 0; i < 4; ++i) {
      int idx = tid + i * 256;
      int row = idx >> 3, c4 = (idx & 7) << 2;
      v4bf bv;
      bv[0] = (bf16_t)areg[i].x; bv[1] = (bf16_t)areg[i].y;
      bv[2] = (bf16_t)areg[i].z; bv[3] = (bf16_t)areg[i].w;
      *(v4bf*)(&As[buf][row * LDA + c4]) = bv;
    }
#pragma unroll
    for (int i = 0; i < 2; ++i) {
      int idx = tid + i * 256;
      int row = idx >> 4, c8 = (idx & 15) << 3;
      *(v8bf*)(&Bs[buf][row * LDB + c8]) = breg[i];
    }
  };

  v8f acc[8];
#pragma unroll
  for (int t = 0; t < 8; ++t) acc[t] = (v8f)0.0f;

  load_tile(0);
  store_tile(0);
  __syncthreads();
  int cur = 0;
  for (int k0 = 0; k0 < DIN; k0 += KC) {
    const bool more = (k0 + KC) < DIN;
    if (more) load_tile(k0 + KC);                   // global loads overlap WMMA
    v16bf a = frag_a(&As[cur][0] + wave * 16 * LDA, lane);
    v16bf bfr[8];
#pragma unroll
    for (int nt = 0; nt < 8; ++nt) bfr[nt] = frag_b(&Bs[cur][0], lane, nt * 16);
#pragma unroll
    for (int nt = 0; nt < 8; ++nt) acc[nt] = wmma_bf16(a, bfr[nt], acc[nt]);
    if (more) {
      store_tile(cur ^ 1);
      __syncthreads();
      cur ^= 1;
    }
  }
#pragma unroll
  for (int nt = 0; nt < 8; ++nt) {
    int col = nt * 16 + (lane & 15);
#pragma unroll
    for (int r = 0; r < 8; ++r) {
      int row = m0 + wave * 16 + r + ((lane >> 4) << 3);
      as_out[((size_t)b * S + row) * KS + col] = (bf16_t)(acc[nt][r] * 0.0078125f); // 1/128
    }
  }
}

// ---------------- kernel 4: out[b] = AS[b] @ SB[b] + bias  (4096 x 1024, K=128) ----------------

__global__ __launch_bounds__(256) void rla_out(const bf16_t* __restrict__ as_in,
                                               const bf16_t* __restrict__ sb,
                                               const float* __restrict__ bias,
                                               float* __restrict__ out) {
  __shared__ __align__(16) bf16_t As[2][128 * LDA];
  __shared__ __align__(16) bf16_t Bs[2][KC * LDB];
  const int tid = threadIdx.x, wave = tid >> 5, lane = tid & 31;
  const int m0 = blockIdx.x * 128;                  // 32
  const int n0 = blockIdx.y * 128;                  // 8
  const int b  = blockIdx.z;                        // 8

  v8bf areg[2], breg[2];
  auto load_tile = [&](int k0) {
#pragma unroll
    for (int i = 0; i < 2; ++i) {                   // A: 128x32 bf16
      int idx = tid + i * 256;
      int row = idx >> 2, c8 = (idx & 3) << 3;
      areg[i] = *(const v8bf*)(as_in + ((size_t)b * S + m0 + row) * KS + k0 + c8);
    }
#pragma unroll
    for (int i = 0; i < 2; ++i) {                   // B: 32x128 bf16
      int idx = tid + i * 256;
      int row = idx >> 4, c8 = (idx & 15) << 3;
      breg[i] = *(const v8bf*)(sb + ((size_t)b * KS + k0 + row) * DOUT + n0 + c8);
    }
  };
  auto store_tile = [&](int buf) {
#pragma unroll
    for (int i = 0; i < 2; ++i) {
      int idx = tid + i * 256;
      int row = idx >> 2, c8 = (idx & 3) << 3;
      *(v8bf*)(&As[buf][row * LDA + c8]) = areg[i];
    }
#pragma unroll
    for (int i = 0; i < 2; ++i) {
      int idx = tid + i * 256;
      int row = idx >> 4, c8 = (idx & 15) << 3;
      *(v8bf*)(&Bs[buf][row * LDB + c8]) = breg[i];
    }
  };

  v8f acc[8];
#pragma unroll
  for (int t = 0; t < 8; ++t) acc[t] = (v8f)0.0f;

  load_tile(0);
  store_tile(0);
  __syncthreads();
  int cur = 0;
#pragma unroll
  for (int k0 = 0; k0 < KS; k0 += KC) {             // 4 chunks
    const bool more = (k0 + KC) < KS;
    if (more) load_tile(k0 + KC);
    v16bf a = frag_a(&As[cur][0] + wave * 16 * LDA, lane);
    v16bf bfr[8];
#pragma unroll
    for (int nt = 0; nt < 8; ++nt) bfr[nt] = frag_b(&Bs[cur][0], lane, nt * 16);
#pragma unroll
    for (int nt = 0; nt < 8; ++nt) acc[nt] = wmma_bf16(a, bfr[nt], acc[nt]);
    if (more) {
      store_tile(cur ^ 1);
      __syncthreads();
      cur ^= 1;
    }
  }
#pragma unroll
  for (int nt = 0; nt < 8; ++nt) {
    int col = n0 + nt * 16 + (lane & 15);
    float bv = bias[col];
#pragma unroll
    for (int r = 0; r < 8; ++r) {
      int row = m0 + wave * 16 + r + ((lane >> 4) << 3);
      out[((size_t)b * S + row) * DOUT + col] = acc[nt][r] + bv;
    }
  }
}

// ---------------- launcher ----------------

extern "C" void kernel_launch(void* const* d_in, const int* in_sizes, int n_in,
                              void* d_out, int out_size, void* d_ws, size_t ws_size,
                              hipStream_t stream) {
  const float* x    = (const float*)d_in[0];   // [8,4096,1024]
  const float* w    = (const float*)d_in[1];   // [1024,1024]
  const float* bias = (const float*)d_in[2];   // [1024]
  float*       out  = (float*)d_out;           // [8,4096,1024]

  // workspace: Sk (2MB) | SB (2MB) | AS (8MB) = 12MB bf16
  bf16_t* sk = (bf16_t*)d_ws;
  bf16_t* sb = sk + (size_t)NB * DIN * KS;
  bf16_t* as = sb + (size_t)NB * KS * DOUT;

  rla_gen_sketch<<<dim3((NB * DIN * KS / 64) / 256), dim3(256), 0, stream>>>(sk);
  rla_sb<<<dim3(DOUT / 128, NB), dim3(256), 0, stream>>>(sk, w, sb);
  rla_as<<<dim3(NB * (S / 128)), dim3(256), 0, stream>>>(x, sk, as);
  rla_out<<<dim3(S / 128, DOUT / 128, NB), dim3(256), 0, stream>>>(as, sb, bias, out);
}
